// HungarianMatcher_35390530519695
// MI455X (gfx1250) — matrix operand order repaired and checked by reading, not compiled, not run on hardware
//
#include <hip/hip_runtime.h>
#include <hip/hip_bf16.h>

#define BS    32
#define NQ    900
#define NCLS  1203
#define NT    64
#define QT    16
#define ROWP  1216   // padded LDS row stride (floats, 16B aligned, = 38*32)
#define KPAD  1216   // 38 * 32 K iterations

typedef __attribute__((ext_vector_type(16))) __bf16 v16bf;
typedef __attribute__((ext_vector_type(8)))  float  v8f;
typedef __attribute__((ext_vector_type(8)))  int    v8i;

extern __shared__ char smem_raw[];

// ---------------------------------------------------------------------------
// Kernel 1: one wave (32 threads) per 16-query tile.
//   - async-stage 16x1203 fp32 logits tile into LDS (HBM read exactly once)
//   - max-only pass, then exp + sum(exp) fused into the WMMA A-operand build
//   - cost = -(exp(x-max) @ OneHot(labels)) / Z  via v_wmma_f32_16x16x32_bf16
//   - one-hot B built branch-free as 8 packed dwords (cmp+cndmask per pair)
//   - store transposed costT[b][t][q] so Hungarian row scans are coalesced
// ---------------------------------------------------------------------------
__global__ __launch_bounds__(32)
void cost_tile_kernel(const float* __restrict__ logits,
                      const int*   __restrict__ labels,
                      float*       __restrict__ costT)
{
    float* ldsf = (float*)smem_raw;            // QT*ROWP staged logits (padded)
    float* smax = ldsf + QT * ROWP;            // per-row max
    float* sinv = smax + QT;                   // per-row 1/sum(exp)

    const int lane = threadIdx.x;
    const int q0   = blockIdx.x * QT;
    const int b    = blockIdx.y;

    const int validRows = (NQ - q0 < QT) ? (NQ - q0) : QT;
    const float* gbase  = logits + ((size_t)b * NQ + q0) * NCLS;

    // zero pad rows (partial last tile) and pad columns 1203..1215
    for (int r = validRows; r < QT; ++r)
        for (int w = lane; w < ROWP; w += 32) ldsf[r * ROWP + w] = 0.0f;
    if (lane < ROWP - NCLS)
        for (int r = 0; r < validRows; ++r) ldsf[r * ROWP + NCLS + lane] = 0.0f;

#if __has_builtin(__builtin_amdgcn_global_load_async_to_lds_b32)
    {   // CDNA5 async global->LDS DMA path (ASYNCcnt tracked)
        typedef __attribute__((address_space(1))) int gint;   // "__device__ int *"
        typedef __attribute__((address_space(3))) int lint;   // "__shared__ int *"
        for (int r = 0; r < validRows; ++r) {
            gint* g = (gint*)(gbase + (size_t)r * NCLS);
            lint* l = (lint*)(ldsf + r * ROWP);
            for (int k = lane; k < NCLS; k += 32)
                __builtin_amdgcn_global_load_async_to_lds_b32(g + k, l + k, 0, 0);
        }
#if __has_builtin(__builtin_amdgcn_s_wait_asynccnt)
        __builtin_amdgcn_s_wait_asynccnt(0);
#else
        asm volatile("s_wait_asynccnt 0x0" ::: "memory");
#endif
    }
#else
    for (int r = 0; r < validRows; ++r)
        for (int k = lane; k < NCLS; k += 32)
            ldsf[r * ROWP + k] = gbase[(size_t)r * NCLS + k];
#endif
    __syncthreads();

    // ---- per-row max (cheap pass, no exp), wave32 butterfly ----
    for (int r = 0; r < QT; ++r) {
        const float* row = ldsf + r * ROWP;
        float mx = -__builtin_inff();
        for (int k = lane; k < NCLS; k += 32) mx = fmaxf(mx, row[k]);
#pragma unroll
        for (int off = 16; off > 0; off >>= 1)
            mx = fmaxf(mx, __shfl_xor(mx, off, 32));
        if (lane == 0) smax[r] = mx;
    }
    __syncthreads();

    // ---- WMMA: A = exp(logits - max) [16 x 1216 bf16], B = one-hot labels ----
    const int m    = lane & 15;               // A row
    const int koff = (lane < 16) ? 0 : 8;     // A lane K base (16-bit A layout)
    const int boff = (lane < 16) ? 0 : 8;     // B lane K base (16-bit B layout)
    const float rmax  = smax[m];
    const float* arow = ldsf + m * ROWP;      // 16B aligned (ROWP*4 % 16 == 0)

    int lab[4];
#pragma unroll
    for (int t = 0; t < 4; ++t) lab[t] = labels[b * NT + t * 16 + m];

    v8f acc[4];
#pragma unroll
    for (int t = 0; t < 4; ++t) acc[t] = (v8f){0.f,0.f,0.f,0.f,0.f,0.f,0.f,0.f};

    float ssum = 0.0f;                        // fused sum(exp) for row m (half)

    for (int kk = 0; kk < KPAD; kk += 32) {
        const int k0 = kk + koff;             // multiple of 8 -> 32B aligned
        const int k1 = k0 + 16;
        float4 L0 = *(const float4*)(arow + k0);
        float4 L1 = *(const float4*)(arow + k0 + 4);
        float4 H0 = *(const float4*)(arow + k1);
        float4 H1 = *(const float4*)(arow + k1 + 4);
        float xs[16] = { L0.x, L0.y, L0.z, L0.w, L1.x, L1.y, L1.z, L1.w,
                         H0.x, H0.y, H0.z, H0.w, H1.x, H1.y, H1.z, H1.w };
        v16bf am;
#pragma unroll
        for (int e = 0; e < 8; ++e) {
            float x0 = (k0 + e < NCLS) ? __expf(xs[e]     - rmax) : 0.0f;
            float x1 = (k1 + e < NCLS) ? __expf(xs[8 + e] - rmax) : 0.0f;
            am[e]     = (__bf16)x0;
            am[8 + e] = (__bf16)x1;
            ssum += x0 + x1;
        }
        const int kkb = kk + boff;            // this lane's B K-range base
#pragma unroll
        for (int t = 0; t < 4; ++t) {
            // one-hot B as 8 packed dwords: element s is 1.0bf16 iff
            // lab == kkb + s (s<8) or kkb + 8 + s (s>=8).
            const int rel     = lab[t] - kkb;
            const int relHalf = (rel >= 0 && rel < 8) ? (rel >> 1) : (rel >= 16 && rel < 24) ? ((rel - 8) >> 1) : -1;
            const int val01   = (rel & 1) ? 0x3F800000 : 0x00003F80;
            v8i bi;
#pragma unroll
            for (int p = 0; p < 8; ++p) bi[p] = (relHalf == p) ? val01 : 0;
            v16bf bm = __builtin_bit_cast(v16bf, bi);
            acc[t] = __builtin_amdgcn_wmma_f32_16x16x32_bf16(
                         false, am, false, bm, (short)0, acc[t], false, false);
        }
    }

    // combine lane-pair partial sums -> row sum, publish 1/Z
    ssum += __shfl_xor(ssum, 16, 32);
    if (lane < 16) sinv[m] = 1.0f / ssum;
    __syncthreads();

    // ---- writeout: costT[b][tgt][q] = -acc * (1/Z_row) ----
    const int colN  = lane & 15;
    const int rbase = (lane < 16) ? 0 : 8;
    float invv[8];
#pragma unroll
    for (int vv = 0; vv < 8; ++vv) invv[vv] = sinv[rbase + vv];

#pragma unroll
    for (int t = 0; t < 4; ++t) {
        const int tgt = t * 16 + colN;
        float* dst = costT + ((size_t)b * NT + tgt) * NQ;
#pragma unroll
        for (int vv = 0; vv < 8; ++vv) {
            const int q = q0 + rbase + vv;
            if (q < NQ) dst[q] = -acc[t][vv] * invv[vv];
        }
    }
}

// ---------------------------------------------------------------------------
// Kernel 2: Jonker-Volgenant assignment, one block per batch image.
// a[i][j] = costT[b][i][j], i = target (64 rows), j = query (900 cols).
// fp64 to match the numpy reference; all state in LDS.
// ---------------------------------------------------------------------------
#define HBLK 256

__global__ __launch_bounds__(HBLK)
void hungarian_kernel(const float* __restrict__ costT,
                      float*       __restrict__ out)
{
    __shared__ double u[NT + 1];
    __shared__ double v[NQ + 1];
    __shared__ double minv[NQ + 1];
    __shared__ int    p[NQ + 1];
    __shared__ int    way[NQ + 1];
    __shared__ unsigned char used[NQ + 1];
    __shared__ double red_val[HBLK];
    __shared__ int    red_idx[HBLK];
    __shared__ int    sj0;

    const int tid = threadIdx.x;
    const int b   = blockIdx.x;
    const float* A = costT + (size_t)b * NT * NQ;
    const double INF = __builtin_inf();

    for (int j = tid; j <= NQ; j += HBLK) { v[j] = 0.0; p[j] = 0; way[j] = 0; }
    for (int i = tid; i <= NT; i += HBLK) u[i] = 0.0;
    __syncthreads();

    for (int i = 1; i <= NT; ++i) {
        if (tid == 0) { p[0] = i; sj0 = 0; }
        for (int j = tid; j <= NQ; j += HBLK) { minv[j] = INF; used[j] = 0; }
        __syncthreads();

        while (true) {
            const int j0 = sj0;
            if (tid == 0) used[j0] = 1;
            __syncthreads();

            const int    i0  = p[j0];
            const double ui0 = u[i0];

            double best = INF;
            int    bidx = NQ + 1;
            for (int j = 1 + tid; j <= NQ; j += HBLK) {
                if (!used[j]) {
                    double cur = (double)A[(size_t)(i0 - 1) * NQ + (j - 1)] - ui0 - v[j];
                    if (cur < minv[j]) { minv[j] = cur; way[j] = j0; }
                    double mv = minv[j];
                    if (mv < best || (mv == best && j < bidx)) { best = mv; bidx = j; }
                }
            }
            red_val[tid] = best;
            red_idx[tid] = bidx;
            __syncthreads();
            for (int s = HBLK / 2; s > 0; s >>= 1) {
                if (tid < s) {
                    double ov = red_val[tid + s];
                    int    oi = red_idx[tid + s];
                    if (ov < red_val[tid] || (ov == red_val[tid] && oi < red_idx[tid])) {
                        red_val[tid] = ov; red_idx[tid] = oi;
                    }
                }
                __syncthreads();
            }
            const int    j1    = red_idx[0];
            const double delta = red_val[0];

            // used columns carry distinct rows p[j] -> no atomics needed
            for (int j = tid; j <= NQ; j += HBLK) {
                if (used[j]) { u[p[j]] += delta; v[j] -= delta; }
                else         { minv[j] -= delta; }
            }
            if (tid == 0) sj0 = j1;
            __syncthreads();
            if (p[j1] == 0) break;
        }

        if (tid == 0) {   // augment along alternating path
            int j0 = sj0;
            while (j0 != 0) { int j1 = way[j0]; p[j0] = p[j1]; j0 = j1; }
        }
        __syncthreads();
    }

    if (tid == 0) {       // emit (query ascending, matched target) as floats
        int idx = 0;
        for (int j = 1; j <= NQ; ++j) {
            if (p[j] != 0) {
                out[b * NT + idx]            = (float)(j - 1);   // rows: query idx
                out[BS * NT + b * NT + idx]  = (float)(p[j] - 1);// cols: target idx
                ++idx;
            }
        }
    }
}

// ---------------------------------------------------------------------------
extern "C" void kernel_launch(void* const* d_in, const int* in_sizes, int n_in,
                              void* d_out, int out_size, void* d_ws, size_t ws_size,
                              hipStream_t stream)
{
    const float* logits = (const float*)d_in[0];   // [32,900,1203] f32
    const int*   labels = (const int*)d_in[1];     // [32,64] int
    float* costT = (float*)d_ws;                   // [32][64][900] f32 = 7.37 MB

    dim3 g1((NQ + QT - 1) / QT, BS);               // 57 x 32 tiles, 1 wave each
    size_t smem1 = (size_t)(QT * ROWP + 2 * QT) * sizeof(float);  // ~78 KB LDS
    cost_tile_kernel<<<g1, 32, smem1, stream>>>(logits, labels, costT);

    hungarian_kernel<<<BS, HBLK, 0, stream>>>(costT, (float*)d_out);
}
